// DLRM_Net_27582279974968
// MI455X (gfx1250) — matrix operand order repaired and checked by reading, not compiled, not run on hardware
//
#include <hip/hip_runtime.h>
#include <cstdint>
#include <cstddef>

// ---------------------------------------------------------------------------
// DLRM forward for MI455X (gfx1250, wave32, WMMA).
// bf16 activations/weights, fp32 accumulation via v_wmma_f32_16x16x32_bf16.
// ---------------------------------------------------------------------------

typedef __bf16 bf16;
typedef __attribute__((ext_vector_type(16))) __bf16 v16bf;
typedef __attribute__((ext_vector_type(8)))  float  v8f;

#define BATCH    16384
#define EMB      128
#define VOCAB    100000
#define NSPARSE  26

// ------------------------------- prep kernels ------------------------------

// dense (B x 13 f32) -> (B x 32 bf16), K padded with zeros
__global__ __launch_bounds__(256) void dense_pad_kernel(
    const float* __restrict__ D, bf16* __restrict__ Dp)
{
    int idx = blockIdx.x * 256 + threadIdx.x;       // over B*32
    int b = idx >> 5, k = idx & 31;
    Dp[idx] = (k < 13) ? (bf16)D[b * 13 + k] : (bf16)0.f;
}

// W (K x N f32, row major) -> WT (N x Kp bf16), transposed + K padded
__global__ __launch_bounds__(256) void wcvt_kernel(
    const float* __restrict__ W, bf16* __restrict__ WT, int K, int N, int Kp)
{
    int idx = blockIdx.x * 256 + threadIdx.x;
    if (idx >= N * Kp) return;
    int n = idx / Kp, k = idx - n * Kp;
    WT[(size_t)n * Kp + k] = (k < K) ? (bf16)W[(size_t)k * N + n] : (bf16)0.f;
}

// ------------------------------- WMMA GEMM ---------------------------------
// C(M x N, bf16) = relu(A(M x K, bf16) @ W + bias), W given as WT (N x K bf16).
// Block: 256 threads = 8 waves; tile 128x128; each wave computes 32x64
// as a 2x4 grid of 16x16 WMMA fragments; K-step 32. 128x128 tiling halves
// redundant A/B panel traffic vs 64x64 (the dominant byte cost of the top
// MLP on a 23.3 TB/s part).
__global__ __launch_bounds__(256) void gemm_bias_relu_bf16(
    const bf16* __restrict__ A, const bf16* __restrict__ WT,
    const float* __restrict__ bias, bf16* __restrict__ C,
    int M, int N, int K)
{
    __shared__ bf16 ldsA[128 * 32];
    __shared__ bf16 ldsB[128 * 32];    // n-major: [n][k]

    const int tid  = threadIdx.x;
    const int lane = tid & 31;
    const int wid  = tid >> 5;
    const int wm   = (wid >> 1) * 32;  // wave row offset in tile (0..96)
    const int wn   = (wid & 1)  * 64;  // wave col offset in tile (0/64)
    const int lm   = lane & 15;
    const int lh   = lane >> 4;        // lane half (0/1)
    const int m0   = blockIdx.y * 128;
    const int n0   = blockIdx.x * 128;

    v8f acc[2][4] = {};

    for (int k0 = 0; k0 < K; k0 += 32) {
        // Cooperative tile load, 2 bf16 per u32, fully coalesced:
        // 2048 u32 per tile, 8 u32 per thread.
#pragma unroll
        for (int i = 0; i < 8; ++i) {
            int idx = tid * 8 + i;
            int r = idx >> 4, c = idx & 15;   // 16 u32 per row of 32 bf16
            ((uint32_t*)ldsA)[idx] =
                *((const uint32_t*)(A  + (size_t)(m0 + r) * K + k0) + c);
            ((uint32_t*)ldsB)[idx] =
                *((const uint32_t*)(WT + (size_t)(n0 + r) * K + k0) + c);
        }
        // Prefetch next K-step panels into cache (global_prefetch_b8).
        if (k0 + 32 < K) {
            int r = tid >> 1, h = tid & 1;    // 128 rows x 2 halves
            __builtin_prefetch(A  + (size_t)(m0 + r) * K + k0 + 32 + h * 16, 0, 3);
            __builtin_prefetch(WT + (size_t)(n0 + r) * K + k0 + 32 + h * 16, 0, 3);
        }
        __syncthreads();

        // Fragment loads (16-bit, wave32), contiguous K order k = 16*lh + i, same in A and B:
        //   A: lane holds row lm, contiguous K half selected by lane half.
        //   B: lane holds column lm, contiguous K half selected by lane half.
        v16bf af[2], bfr[4];
#pragma unroll
        for (int f = 0; f < 2; ++f) {
            const bf16* pa = &ldsA[(wm + f * 16 + lm) * 32];
#pragma unroll
            for (int i = 0; i < 16; ++i) af[f][i] = pa[lh * 16 + i];
        }
#pragma unroll
        for (int f = 0; f < 4; ++f) {
            const bf16* pb = &ldsB[(wn + f * 16 + lm) * 32 + lh * 16];
#pragma unroll
            for (int i = 0; i < 16; ++i) bfr[f][i] = pb[i];
        }
#pragma unroll
        for (int fm = 0; fm < 2; ++fm)
#pragma unroll
            for (int fn = 0; fn < 4; ++fn)
                acc[fm][fn] = __builtin_amdgcn_wmma_f32_16x16x32_bf16(
                    false, af[fm], false, bfr[fn], (short)0, acc[fm][fn],
                    false, false);
        __syncthreads();
    }

    // Epilogue: bias + ReLU + bf16 store. C/D layout: VGPR r -> row r + 8*lh.
#pragma unroll
    for (int fm = 0; fm < 2; ++fm)
#pragma unroll
        for (int fn = 0; fn < 4; ++fn)
#pragma unroll
            for (int r = 0; r < 8; ++r) {
                int row = m0 + wm + fm * 16 + r + lh * 8;
                int col = n0 + wn + fn * 16 + lm;
                float v = acc[fm][fn][r] + bias[col];
                v = v > 0.f ? v : 0.f;
                C[(size_t)row * N + col] = (bf16)v;
            }
}

// --------------------------- gather + interaction --------------------------
// One wave per batch row. T = [x; emb_0..emb_25; 0-pad] (32 x 128 bf16) in
// LDS; Z = T @ T^T via WMMA (B(k,n) = T(n,k), so B fragments read T rows);
// output row of top-MLP input: [x (128) | triu(Z) (351) | 0 (33)] bf16.
__global__ __launch_bounds__(128) void interact_kernel(
    const bf16* __restrict__ xbf,      // B x 128 (bottom-MLP output)
    const int*  __restrict__ indices,  // 26 x B
    const float* __restrict__ emb,     // 26 x VOCAB x 128
    bf16* __restrict__ topin)          // B x 512
{
    __shared__ bf16  T[4][32 * 128];
    __shared__ float Cm[4][32 * 32];

    const int lane = threadIdx.x & 31;
    const int w    = threadIdx.x >> 5;
    const int b    = blockIdx.x * 4 + w;
    const int lm   = lane & 15;
    const int lh   = lane >> 4;
    bf16* Tw = T[w];

    // Row 0: x (already bf16)
    {
        const bf16* src = xbf + (size_t)b * EMB;
#pragma unroll
        for (int i = 0; i < 4; ++i) Tw[lane * 4 + i] = src[lane * 4 + i];
    }
    // Rows 1..26: gathered embeddings (f32 -> bf16); each wave reads a
    // contiguous 512B row -> fully coalesced gather.
    for (int f = 0; f < NSPARSE; ++f) {
        int idx = indices[f * BATCH + b];
        const float* src = emb + ((size_t)f * VOCAB + idx) * EMB;
#pragma unroll
        for (int i = 0; i < 4; ++i)
            Tw[(f + 1) * EMB + lane * 4 + i] = (bf16)src[lane * 4 + i];
    }
    // Rows 27..31: zero pad
#pragma unroll
    for (int r = 27; r < 32; ++r)
#pragma unroll
        for (int i = 0; i < 4; ++i) Tw[r * EMB + lane * 4 + i] = (bf16)0.f;
    __syncthreads();

    // Z = T @ T^T : 2x2 fragment tiles, K = 128 in 4 steps of 32.
    v8f acc[2][2] = {};
    for (int k0 = 0; k0 < EMB; k0 += 32) {
        v16bf af[2], bfr[2];
#pragma unroll
        for (int f = 0; f < 2; ++f) {
            const bf16* pa = &Tw[(f * 16 + lm) * EMB + k0];
#pragma unroll
            for (int i = 0; i < 16; ++i) af[f][i] = pa[lh * 16 + i];
            const bf16* pb = &Tw[(f * 16 + lm) * EMB + k0 + lh * 16];
#pragma unroll
            for (int i = 0; i < 16; ++i) bfr[f][i] = pb[i];
        }
#pragma unroll
        for (int fm = 0; fm < 2; ++fm)
#pragma unroll
            for (int fn = 0; fn < 2; ++fn)
                acc[fm][fn] = __builtin_amdgcn_wmma_f32_16x16x32_bf16(
                    false, af[fm], false, bfr[fn], (short)0, acc[fm][fn],
                    false, false);
    }

    // Spill Z (32x32 f32) to LDS for the triangular extraction.
#pragma unroll
    for (int fm = 0; fm < 2; ++fm)
#pragma unroll
        for (int fn = 0; fn < 2; ++fn)
#pragma unroll
            for (int r = 0; r < 8; ++r) {
                int m = fm * 16 + r + lh * 8;
                int n = fn * 16 + lm;
                Cm[w][m * 32 + n] = acc[fm][fn][r];
            }
    __syncthreads();

    bf16* outp = topin + (size_t)b * 512;
    // cols [0,128): x
#pragma unroll
    for (int i = 0; i < 4; ++i) outp[lane * 4 + i] = Tw[lane * 4 + i];
    // cols [128,479): triu(Z, k=1) over 27x27; cols [479,512): zero
    for (int t = lane; t < 384; t += 32) {
        bf16 v = (bf16)0.f;
        if (t < 351) {
            int i = 0, t2 = t;
            while (t2 >= 26 - i) { t2 -= 26 - i; ++i; }
            int j = i + 1 + t2;
            v = (bf16)Cm[w][i * 32 + j];
        }
        outp[128 + t] = v;
    }
}

// ------------------------------ final layer --------------------------------
// 256 -> 1 dot + bias + sigmoid, one wave per batch row, fp32.
__global__ __launch_bounds__(256) void final_kernel(
    const bf16* __restrict__ A, const float* __restrict__ w4,
    const float* __restrict__ b4, float* __restrict__ out)
{
    int lane = threadIdx.x & 31;
    int wv   = threadIdx.x >> 5;
    int b    = blockIdx.x * 8 + wv;
    const bf16* row = A + (size_t)b * 256;
    float s = 0.f;
#pragma unroll
    for (int i = 0; i < 8; ++i)
        s += (float)row[lane * 8 + i] * w4[lane * 8 + i];
#pragma unroll
    for (int off = 16; off >= 1; off >>= 1) s += __shfl_xor(s, off, 32);
    if (lane == 0) out[b] = 1.f / (1.f + __expf(-(s + b4[0])));
}

// ------------------------------- launcher ----------------------------------

extern "C" void kernel_launch(void* const* d_in, const int* in_sizes, int n_in,
                              void* d_out, int out_size, void* d_ws, size_t ws_size,
                              hipStream_t stream) {
    const float* dense   = (const float*)d_in[0];
    const int*   indices = (const int*)  d_in[1];
    const float* emb     = (const float*)d_in[2];
    const float* bw0 = (const float*)d_in[3];  const float* bb0 = (const float*)d_in[4];
    const float* bw1 = (const float*)d_in[5];  const float* bb1 = (const float*)d_in[6];
    const float* bw2 = (const float*)d_in[7];  const float* bb2 = (const float*)d_in[8];
    const float* tw0 = (const float*)d_in[9];  const float* tb0 = (const float*)d_in[10];
    const float* tw1 = (const float*)d_in[11]; const float* tb1 = (const float*)d_in[12];
    const float* tw2 = (const float*)d_in[13]; const float* tb2 = (const float*)d_in[14];
    const float* tw3 = (const float*)d_in[15]; const float* tb3 = (const float*)d_in[16];
    const float* tw4 = (const float*)d_in[17]; const float* tb4 = (const float*)d_in[18];
    float* out = (float*)d_out;

    // Workspace carving (all bf16), ~90 MB total.
    char* p = (char*)d_ws;
    auto alloc = [&](size_t bytes) -> void* {
        void* r = (void*)p; p += (bytes + 255) & ~(size_t)255; return r;
    };
    bf16* densePad = (bf16*)alloc((size_t)BATCH * 32 * 2);
    bf16* wt_b0 = (bf16*)alloc((size_t)512  * 32   * 2);
    bf16* wt_b1 = (bf16*)alloc((size_t)256  * 512  * 2);
    bf16* wt_b2 = (bf16*)alloc((size_t)128  * 256  * 2);
    bf16* wt_t0 = (bf16*)alloc((size_t)1024 * 512  * 2);
    bf16* wt_t1 = (bf16*)alloc((size_t)1024 * 1024 * 2);
    bf16* wt_t2 = (bf16*)alloc((size_t)512  * 1024 * 2);
    bf16* wt_t3 = (bf16*)alloc((size_t)256  * 512  * 2);
    bf16* act_x  = (bf16*)alloc((size_t)BATCH * 128  * 2);  // bottom output
    bf16* top_in = (bf16*)alloc((size_t)BATCH * 512  * 2);  // 479 padded
    bf16* bufA   = (bf16*)alloc((size_t)BATCH * 1024 * 2);  // ping
    bf16* bufB   = (bf16*)alloc((size_t)BATCH * 1024 * 2);  // pong

    // --- prep: pad/convert dense, transpose+convert weights -----------------
    dense_pad_kernel<<<(BATCH * 32) / 256, 256, 0, stream>>>(dense, densePad);
    auto wcvt = [&](const float* W, bf16* WT, int K, int N, int Kp) {
        int tot = N * Kp;
        wcvt_kernel<<<(tot + 255) / 256, 256, 0, stream>>>(W, WT, K, N, Kp);
    };
    wcvt(bw0,  wt_b0,  13,  512,  32);
    wcvt(bw1,  wt_b1,  512, 256,  512);
    wcvt(bw2,  wt_b2,  256, 128,  256);
    wcvt(tw0,  wt_t0,  479, 1024, 512);
    wcvt(tw1,  wt_t1,  1024,1024, 1024);
    wcvt(tw2,  wt_t2,  1024,512,  1024);
    wcvt(tw3,  wt_t3,  512, 256,  512);

    auto gemm = [&](const bf16* A, const bf16* WT, const float* bias, bf16* C,
                    int M, int N, int K) {
        dim3 grid(N / 128, M / 128);
        gemm_bias_relu_bf16<<<grid, 256, 0, stream>>>(A, WT, bias, C, M, N, K);
    };

    // --- bottom MLP ---------------------------------------------------------
    gemm(densePad, wt_b0, bb0, bufA,  BATCH, 512, 32);
    gemm(bufA,     wt_b1, bb1, bufB,  BATCH, 256, 512);
    gemm(bufB,     wt_b2, bb2, act_x, BATCH, 128, 256);

    // --- embedding gather + pairwise interaction ----------------------------
    interact_kernel<<<BATCH / 4, 128, 0, stream>>>(act_x, indices, emb, top_in);

    // --- top MLP ------------------------------------------------------------
    gemm(top_in, wt_t0, tb0, bufA, BATCH, 1024, 512);
    gemm(bufA,   wt_t1, tb1, bufB, BATCH, 1024, 1024);
    gemm(bufB,   wt_t2, tb2, bufA, BATCH, 512,  1024);
    gemm(bufA,   wt_t3, tb3, bufB, BATCH, 256,  512);

    // --- final 256 -> 1 + sigmoid ------------------------------------------
    final_kernel<<<BATCH / 8, 256, 0, stream>>>(bufB, tw4, tb4, out);

    (void)in_sizes; (void)n_in; (void)out_size; (void)ws_size;
}
